// SelfAttention_67680094650526
// MI455X (gfx1250) — matrix operand order repaired and checked by reading, not compiled, not run on hardware
//
#include <hip/hip_runtime.h>

typedef __attribute__((ext_vector_type(16))) _Float16 v16h;
typedef __attribute__((ext_vector_type(8)))  float    v8f;

#define BB 4
#define TT 4096
#define DD 4096
#define HH 128
#define BT (BB*TT)

static __device__ __forceinline__ v8f wmma_f16(v16h a, v16h b, v8f c) {
  // D = A(16x32 f16) * B(32x16 f16) + C(16x16 f32)
  return __builtin_amdgcn_wmma_f32_16x16x32_f16(false, a, false, b, (short)0, c,
                                                false, false);
}

// ---------------------------------------------------------------------------
// Kernel 0: transpose+convert one (D,H) f32 weight matrix to (H,D) f16 so that
// WMMA B-fragments become two contiguous 16-byte loads per lane.
// ---------------------------------------------------------------------------
__global__ __launch_bounds__(256) void wt_kernel(const float* __restrict__ W,
                                                 _Float16* __restrict__ Wt) {
  int idx = blockIdx.x * blockDim.x + threadIdx.x;
  if (idx >= DD * HH) return;
  int d = idx / HH, h = idx % HH;
  Wt[(size_t)h * DD + d] = (_Float16)W[idx];
}

// ---------------------------------------------------------------------------
// Kernel 1: fused Q/K/V projection. One wave owns a 16-row strip of x and
// computes all 3 matrices x 8 col-tiles, so x (268 MB) is read exactly once.
// Q,K stored row-major f16; V stored transposed (H x BT) f16 for the PV GEMM.
// ---------------------------------------------------------------------------
__global__ __launch_bounds__(128) void proj_kernel(
    const float* __restrict__ x,
    const _Float16* __restrict__ Wqt, const _Float16* __restrict__ Wkt,
    const _Float16* __restrict__ Wvt,
    _Float16* __restrict__ Qh, _Float16* __restrict__ Kh,
    _Float16* __restrict__ Vt)
{
  const int wid = threadIdx.x >> 5, lane = threadIdx.x & 31;
  const int hi = lane >> 4, lx = lane & 15;
  const int rt = blockIdx.x * 4 + wid;                 // 1024 row tiles of 16
  const float* xrow = x + (size_t)(rt * 16 + lx) * DD; // A row for this lane

  v8f acc[3][8];
#pragma unroll
  for (int m = 0; m < 3; ++m)
#pragma unroll
    for (int c = 0; c < 8; ++c) acc[m][c] = (v8f){0.f,0.f,0.f,0.f,0.f,0.f,0.f,0.f};

  const _Float16* Wt[3] = {Wqt, Wkt, Wvt};

  for (int k0 = 0; k0 < DD; k0 += 32) {
    // A fragment: lane L -> row L%16; K runs [8*hi, 8*hi+8) and [16+8*hi, ..)
    const float* ap = xrow + k0 + 8 * hi;
    float4 a0 = *(const float4*)(ap);
    float4 a1 = *(const float4*)(ap + 4);
    float4 a2 = *(const float4*)(ap + 16);
    float4 a3 = *(const float4*)(ap + 20);
    v16h a;
    a[0]=(_Float16)a0.x; a[1]=(_Float16)a0.y; a[2]=(_Float16)a0.z; a[3]=(_Float16)a0.w;
    a[4]=(_Float16)a1.x; a[5]=(_Float16)a1.y; a[6]=(_Float16)a1.z; a[7]=(_Float16)a1.w;
    a[8]=(_Float16)a2.x; a[9]=(_Float16)a2.y; a[10]=(_Float16)a2.z; a[11]=(_Float16)a2.w;
    a[12]=(_Float16)a3.x; a[13]=(_Float16)a3.y; a[14]=(_Float16)a3.z; a[15]=(_Float16)a3.w;

#pragma unroll
    for (int m = 0; m < 3; ++m) {
#pragma unroll
      for (int c = 0; c < 8; ++c) {
        // B fragment: lane L -> col L%16; K runs [8*hi, 8*hi+8) and [16+8*hi, ..)
        const _Float16* wp = Wt[m] + (size_t)(c * 16 + lx) * DD + k0 + 8 * hi;
        union { v16h v; uint4 u[2]; } bf;
        bf.u[0] = *(const uint4*)(wp);
        bf.u[1] = *(const uint4*)(wp + 16);
        acc[m][c] = wmma_f16(a, bf.v, acc[m][c]);
      }
    }
  }

  // C/D tile: lane L, VGPR r -> row r+8*hi, col lx (within tile)
#pragma unroll
  for (int c = 0; c < 8; ++c) {
#pragma unroll
    for (int r = 0; r < 8; ++r) {
      int row = rt * 16 + r + 8 * hi;
      int col = c * 16 + lx;
      Qh[(size_t)row * HH + col] = (_Float16)acc[0][c][r];
      Kh[(size_t)row * HH + col] = (_Float16)acc[1][c][r];
      Vt[(size_t)col * BT + row] = (_Float16)acc[2][c][r];   // transposed V
    }
  }
}

// ---------------------------------------------------------------------------
// Kernel 2: flash attention. One wave owns a 16-query tile; streams keys in
// blocks of 32 with online softmax. Per-batch K/V is 2 MB -> L2 resident.
// ---------------------------------------------------------------------------
__global__ __launch_bounds__(128) void attn_kernel(
    const _Float16* __restrict__ Qh, const _Float16* __restrict__ Kh,
    const _Float16* __restrict__ Vt, float* __restrict__ out)
{
  __shared__ _Float16 plds[4][16 * 32];  // per-wave P tile (D-layout -> A-layout bounce)
  const int wid = threadIdx.x >> 5, lane = threadIdx.x & 31;
  const int hi = lane >> 4, lx = lane & 15;
  const int qt = blockIdx.x * 4 + wid;       // 1024 query tiles
  const int batch = qt / (TT / 16);
  const int qrow0 = qt * 16;                 // global row in (B*T)
  const int keybase = batch * TT;

  // Preload Q A-fragments for the 4 H-chunks of 32 (loop invariant).
  v16h qf[4];
  {
    const _Float16* qp = Qh + (size_t)(qrow0 + lx) * HH + 8 * hi;
#pragma unroll
    for (int kk = 0; kk < 4; ++kk) {
      union { v16h v; uint4 u[2]; } t;
      t.u[0] = *(const uint4*)(qp + kk * 32);
      t.u[1] = *(const uint4*)(qp + kk * 32 + 16);
      qf[kk] = t.v;
    }
  }

  v8f o[8];
#pragma unroll
  for (int c = 0; c < 8; ++c) o[c] = (v8f){0.f,0.f,0.f,0.f,0.f,0.f,0.f,0.f};
  float m[8], l[8];
#pragma unroll
  for (int r = 0; r < 8; ++r) { m[r] = -1e30f; l[r] = 0.f; }

  const float scale = 0.08838834764831845f;  // 1/sqrt(128)

  for (int s0 = 0; s0 < TT; s0 += 32) {
    const int key0 = keybase + s0;

    // S = Q * K^T for 32 keys: two 16x16 D tiles, 4 k-steps each.
    v8f st[2];
    st[0] = (v8f){0.f,0.f,0.f,0.f,0.f,0.f,0.f,0.f};
    st[1] = (v8f){0.f,0.f,0.f,0.f,0.f,0.f,0.f,0.f};
#pragma unroll
    for (int t = 0; t < 2; ++t) {
      // B fragment: lane col = key (lx), contraction K over H -> two 8-runs.
      const _Float16* kp = Kh + (size_t)(key0 + 16 * t + lx) * HH + 8 * hi;
#pragma unroll
      for (int kk = 0; kk < 4; ++kk) {
        union { v16h v; uint4 u[2]; } bf;
        bf.u[0] = *(const uint4*)(kp + kk * 32);
        bf.u[1] = *(const uint4*)(kp + kk * 32 + 16);
        st[t] = wmma_f16(qf[kk], bf.v, st[t]);
      }
    }

    // Online softmax. C/D layout: VGPR r holds row r+8*hi, col lx -> row
    // reductions are shfl_xor over the 16-lane half group.
#pragma unroll
    for (int r = 0; r < 8; ++r) {
      float s0v = st[0][r] * scale;
      float s1v = st[1][r] * scale;
      float rm = fmaxf(s0v, s1v);
#pragma unroll
      for (int off = 1; off < 16; off <<= 1) rm = fmaxf(rm, __shfl_xor(rm, off, 32));
      float mn = fmaxf(m[r], rm);
      float alpha = __expf(m[r] - mn);
      m[r] = mn;
      float p0 = __expf(s0v - mn);
      float p1 = __expf(s1v - mn);
      float rs = p0 + p1;
#pragma unroll
      for (int off = 1; off < 16; off <<= 1) rs += __shfl_xor(rs, off, 32);
      l[r] = alpha * l[r] + rs;
      plds[wid][(r + 8 * hi) * 32 + lx]      = (_Float16)p0;
      plds[wid][(r + 8 * hi) * 32 + 16 + lx] = (_Float16)p1;
#pragma unroll
      for (int c = 0; c < 8; ++c) o[c][r] *= alpha;
    }

    // Reload P in A-fragment layout (row = lx, two contiguous 8-col runs).
    union { v16h v; uint4 u[2]; } pf;
    {
      const _Float16* pp = &plds[wid][lx * 32 + 8 * hi];
      pf.u[0] = *(const uint4*)(pp);
      pf.u[1] = *(const uint4*)(pp + 16);
    }

    // O += P(16x32) * V(32xH) across 8 output col tiles; V^T gives contiguous
    // contraction loads.
#pragma unroll
    for (int c = 0; c < 8; ++c) {
      const _Float16* vp = Vt + (size_t)(c * 16 + lx) * BT + key0 + 8 * hi;
      union { v16h v; uint4 u[2]; } bf;
      bf.u[0] = *(const uint4*)(vp);
      bf.u[1] = *(const uint4*)(vp + 16);
      o[c] = wmma_f16(pf.v, bf.v, o[c]);
    }
  }

  // Normalize and write (B,T,H) f32 output.
#pragma unroll
  for (int r = 0; r < 8; ++r) {
    float inv = 1.0f / l[r];
    int row = qrow0 + r + 8 * hi;
#pragma unroll
    for (int c = 0; c < 8; ++c)
      out[(size_t)row * HH + c * 16 + lx] = o[c][r] * inv;
  }
}

extern "C" void kernel_launch(void* const* d_in, const int* in_sizes, int n_in,
                              void* d_out, int out_size, void* d_ws, size_t ws_size,
                              hipStream_t stream) {
  const float* x  = (const float*)d_in[0];
  const float* Wq = (const float*)d_in[1];
  const float* Wk = (const float*)d_in[2];
  const float* Wv = (const float*)d_in[3];
  float* out = (float*)d_out;

  // Workspace: Qh,Kh (BTxH f16), Vt (HxBT f16), 3 transposed f16 weights. 15 MB.
  _Float16* Qh  = (_Float16*)d_ws;
  _Float16* Kh  = Qh  + (size_t)BT * HH;
  _Float16* Vt  = Kh  + (size_t)BT * HH;
  _Float16* Wqt = Vt  + (size_t)BT * HH;
  _Float16* Wkt = Wqt + (size_t)DD * HH;
  _Float16* Wvt = Wkt + (size_t)DD * HH;

  const int wtn = DD * HH;
  wt_kernel<<<(wtn + 255) / 256, 256, 0, stream>>>(Wq, Wqt);
  wt_kernel<<<(wtn + 255) / 256, 256, 0, stream>>>(Wk, Wkt);
  wt_kernel<<<(wtn + 255) / 256, 256, 0, stream>>>(Wv, Wvt);

  proj_kernel<<<BT / 16 / 4, 128, 0, stream>>>(x, Wqt, Wkt, Wvt, Qh, Kh, Vt);
  attn_kernel<<<BT / 16 / 4, 128, 0, stream>>>(Qh, Kh, Vt, out);
}
